// PointNet2_5274219839803
// MI455X (gfx1250) — matrix operand order repaired and checked by reading, not compiled, not run on hardware
//
#include <hip/hip_runtime.h>
#include <hip/hip_bf16.h>

// ---------------- types for CDNA5 WMMA ----------------
typedef __attribute__((ext_vector_type(16))) _Float16 v16h;
typedef __attribute__((ext_vector_type(8)))  _Float16 v8h;
typedef __attribute__((ext_vector_type(8)))  float    v8f;

// ---------------- problem constants ----------------
#define B_   8
#define N_   8192
#define FEAT_ 1
#define M1_  1638      // int(8192*0.2)
#define M2_  409       // int(1638*0.25)
#define KNB  64
#define R1SQ 0.04f     // 0.2^2
#define R2SQ 0.16f     // 0.4^2
#define OUTD 32

#define R3_  (B_*M2_)          // 3272
#define RP3_ 3328              // round up to 64
#define R5_  (B_*M1_)          // 13104
#define RP5_ 13120             // round up to 64
#define R6_  (B_*N_)           // 65536

// ---------------- WMMA helpers ----------------
__device__ __forceinline__ v8f wmma_f16(v16h a, v16h b, v8f c) {
  // D = A(16x32) * B(32x16) + C, f32 accumulate
  return __builtin_amdgcn_wmma_f32_16x16x32_f16(false, a, false, b, (short)0, c, false, false);
}

// Load a 16x32 A-fragment from row-major f16 (LDS or global), row stride ldh halves.
// A layout (16-bit, 16x32): lane<16 holds row=lane, K={kc+0..7, kc+16..23};
//                           lane>=16 holds row=lane-16, K={kc+8..15, kc+24..31}.
__device__ __forceinline__ v16h load_a_frag(const _Float16* base, int ldh, int kc) {
  int lane = threadIdx.x & 31;
  int r = lane & 15;
  const _Float16* p = base + (size_t)r * ldh + kc + ((lane >> 4) << 3);
  v8h lo = *(const v8h*)p;
  v8h hi = *(const v8h*)(p + 16);
  v16h f;
#pragma unroll
  for (int i = 0; i < 8; i++) { f[i] = lo[i]; f[i + 8] = hi[i]; }
  return f;
}

// Load a 32x16 B-fragment from PRE-SWIZZLED f16 weights WT[Cout][KP] (transposed,
// zero-padded K). Lane holds col n = col0+(lane&15); K = {kc+8*(lane>=16)+0..7,
// kc+16+8*(lane>=16)+0..7} => two 16B vector loads.
__device__ __forceinline__ v16h load_bt_frag(const _Float16* WT, int KP, int kc, int col0) {
  int lane = threadIdx.x & 31;
  int n = col0 + (lane & 15);
  int kb = kc + ((lane >> 4) << 3);
  const _Float16* p = WT + (size_t)n * KP + kb;
  v8h lo = *(const v8h*)p;
  v8h hi = *(const v8h*)(p + 16);
  v16h f;
#pragma unroll
  for (int i = 0; i < 8; i++) { f[i] = lo[i]; f[i + 8] = hi[i]; }
  return f;
}

// Store 16x16 C/D tile (v8f) + bias (+relu) as f16 into row-major dst (ld halves).
// C layout: lane holds col=lane&15; VGPR i -> row i + 8*(lane>=16).
__device__ __forceinline__ void store_c_f16(v8f acc, _Float16* dst, int ld, int col0,
                                            const float* bias, bool relu) {
  int lane = threadIdx.x & 31;
  int n = lane & 15, hi = lane >> 4;
  float bv = bias[col0 + n];
#pragma unroll
  for (int i = 0; i < 8; i++) {
    float v = acc[i] + bv;
    if (relu) v = fmaxf(v, 0.0f);
    dst[(size_t)(i + 8 * hi) * ld + col0 + n] = (_Float16)v;
  }
}

__device__ __forceinline__ void store_c_f32(v8f acc, float* dst, int ld, int col0,
                                            const float* bias, bool relu) {
  int lane = threadIdx.x & 31;
  int n = lane & 15, hi = lane >> 4;
  float bv = bias[col0 + n];
#pragma unroll
  for (int i = 0; i < 8; i++) {
    float v = acc[i] + bv;
    if (relu) v = fmaxf(v, 0.0f);
    dst[(size_t)(i + 8 * hi) * ld + col0 + n] = v;
  }
}

// ================= weight pre-swizzle: W[Cin x Cout] f32 -> WT[Cout x KP] f16 =============
__global__ void prep_wt_kernel(const float* __restrict__ W, int Cin, int Cout, int KP,
                               _Float16* __restrict__ WT) {
  int e = blockIdx.x * 256 + threadIdx.x;
  if (e >= Cout * KP) return;
  int n = e / KP, k = e - n * KP;
  WT[e] = (k < Cin) ? (_Float16)W[(size_t)k * Cout + n] : (_Float16)0.0f;
}

// ================= FPS: farthest point sampling =================
// grid = B, block = 256. Writes query positions posq[b, 0..M-1, 3].
__global__ void fps_kernel(const float* __restrict__ pos, int Np, int M,
                           float* __restrict__ posq) {
  __shared__ float dmin[8192];
  __shared__ float rv[256];
  __shared__ int   ri[256];
  __shared__ int   sidx[1664];
  int b = blockIdx.x, tid = threadIdx.x;
  const float* P = pos + (size_t)b * Np * 3;
  for (int i = tid; i < Np; i += 256) dmin[i] = __builtin_inff();
  __syncthreads();
  int last = 0;
  for (int t = 0; t < M; t++) {
    if (tid == 0) sidx[t] = last;
    float lx = P[last * 3 + 0], ly = P[last * 3 + 1], lz = P[last * 3 + 2];
    float bv = -1.0f; int bi = 0;
    for (int i = tid; i < Np; i += 256) {
      float dx = P[i * 3 + 0] - lx, dy = P[i * 3 + 1] - ly, dz = P[i * 3 + 2] - lz;
      float d = dx * dx + dy * dy + dz * dz;
      float nd = fminf(dmin[i], d);
      dmin[i] = nd;
      if (nd > bv) { bv = nd; bi = i; }   // ascending i + strict '>' => lowest index on tie
    }
    rv[tid] = bv; ri[tid] = bi;
    __syncthreads();
    for (int s = 128; s > 0; s >>= 1) {
      if (tid < s) {
        if (rv[tid + s] > rv[tid] || (rv[tid + s] == rv[tid] && ri[tid + s] < ri[tid])) {
          rv[tid] = rv[tid + s]; ri[tid] = ri[tid + s];
        }
      }
      __syncthreads();
    }
    last = ri[0];
    __syncthreads();
  }
  for (int m = tid; m < M; m += 256) {
    int s = sidx[m];
    posq[((size_t)b * M + m) * 3 + 0] = P[s * 3 + 0];
    posq[((size_t)b * M + m) * 3 + 1] = P[s * 3 + 1];
    posq[((size_t)b * M + m) * 3 + 2] = P[s * 3 + 2];
  }
}

// ================= top-64 nearest (radius handled later via mask) =================
// grid = (M, B), block = 256. LDS-resident distance array, 64 x block-argmin.
__global__ void topk_kernel(const float* __restrict__ pos_src, int Np,
                            const float* __restrict__ posq, int M,
                            int* __restrict__ nbr, float* __restrict__ d2o) {
  __shared__ float ds[8192];
  __shared__ float rv[256];
  __shared__ int   ri[256];
  int b = blockIdx.y, m = blockIdx.x, tid = threadIdx.x;
  size_t qb = (size_t)b * M + m;
  const float* P = pos_src + (size_t)b * Np * 3;
  float qx = posq[qb * 3 + 0], qy = posq[qb * 3 + 1], qz = posq[qb * 3 + 2];
  for (int i = tid; i < Np; i += 256) {
    float dx = P[i * 3 + 0] - qx, dy = P[i * 3 + 1] - qy, dz = P[i * 3 + 2] - qz;
    ds[i] = dx * dx + dy * dy + dz * dz;
  }
  __syncthreads();
  for (int k = 0; k < KNB; k++) {
    float bv = __builtin_inff(); int bi = 0x7fffffff;
    for (int i = tid; i < Np; i += 256) {
      float v = ds[i];
      if (v < bv) { bv = v; bi = i; }
    }
    rv[tid] = bv; ri[tid] = bi;
    __syncthreads();
    for (int s = 128; s > 0; s >>= 1) {
      if (tid < s) {
        if (rv[tid + s] < rv[tid] || (rv[tid + s] == rv[tid] && ri[tid + s] < ri[tid])) {
          rv[tid] = rv[tid + s]; ri[tid] = ri[tid + s];
        }
      }
      __syncthreads();
    }
    if (tid == 0) {
      nbr[qb * KNB + k] = ri[0];
      d2o[qb * KNB + k] = rv[0];
      ds[ri[0]] = __builtin_inff();
    }
    __syncthreads();
  }
}

// ================= fused Set-Abstraction MLP (WMMA) =================
// grid = (M, B), block = 128 (4 waves). Per query: gather [64 x Cin] -> f16 LDS,
// layer1 (Cin->32, relu) and layer2 (32->Cout) via v_wmma, masked max over 64 neighbors.
__global__ void sa_fused_kernel(const float* __restrict__ xsrc, const float* __restrict__ pos_src,
                                const float* __restrict__ posq, const int* __restrict__ nbr,
                                const float* __restrict__ d2, float r2, int Np, int M,
                                int F, int CinPad, int lg2Cin, int Cout,
                                const _Float16* __restrict__ WT0, const float* __restrict__ b0,
                                const _Float16* __restrict__ WT1, const float* __restrict__ b1,
                                float* __restrict__ xout) {
  __shared__ _Float16 aT[64 * 64];
  __shared__ _Float16 h1s[64 * 32];
  __shared__ float    h2s[64 * 64];
  int b = blockIdx.y, m = blockIdx.x, tid = threadIdx.x;
  size_t qb = (size_t)b * M + m;
  const int*   nb = nbr + qb * KNB;
  const float* dd = d2 + qb * KNB;
  float qx = posq[qb * 3 + 0], qy = posq[qb * 3 + 1], qz = posq[qb * 3 + 2];
  int total = KNB << lg2Cin;
  for (int e = tid; e < total; e += 128) {
    int row = e >> lg2Cin;
    int c = e & (CinPad - 1);
    int src = nb[row];
    float v = 0.0f;
    if (c < F) {
      v = xsrc[((size_t)b * Np + src) * F + c];
    } else if (c < F + 3) {
      int cc = c - F;
      float q = (cc == 0) ? qx : ((cc == 1) ? qy : qz);
      v = pos_src[((size_t)b * Np + src) * 3 + cc] - q;
    }
    aT[e] = (_Float16)v;
  }
  __syncthreads();
  int w = tid >> 5;
  const _Float16* aw = aT + ((size_t)(w * 16) << lg2Cin);
  _Float16* h1w = h1s + w * 16 * 32;
  // layer 1: Cin -> 32, relu
  for (int ct = 0; ct < 32; ct += 16) {
    v8f acc = {};
    for (int kc = 0; kc < CinPad; kc += 32) {
      v16h af = load_a_frag(aw, CinPad, kc);
      v16h bf = load_bt_frag(WT0, CinPad, kc, ct);
      acc = wmma_f16(af, bf, acc);
    }
    store_c_f16(acc, h1w, 32, ct, b0, true);
  }
  // layer 2: 32 -> Cout, plain, masked into h2s
  int lane = tid & 31;
  int n = lane & 15, hi = lane >> 4;
  for (int ct = 0; ct < Cout; ct += 16) {
    v8f acc = {};
    v16h af = load_a_frag(h1w, 32, 0);
    v16h bf = load_bt_frag(WT1, 32, 0, ct);
    acc = wmma_f16(af, bf, acc);
    float bv = b1[ct + n];
#pragma unroll
    for (int i = 0; i < 8; i++) {
      int rloc = w * 16 + i + 8 * hi;
      float v = acc[i] + bv;
      if (!(dd[rloc] <= r2)) v = -__builtin_inff();
      h2s[(size_t)rloc * Cout + ct + n] = v;
    }
  }
  __syncthreads();
  for (int c = tid; c < Cout; c += 128) {
    float mv = -__builtin_inff();
    for (int r = 0; r < KNB; r++) mv = fmaxf(mv, h2s[(size_t)r * Cout + c]);
    xout[qb * Cout + c] = mv;
  }
}

// ================= generic 2-layer MLP (WMMA), f16 in, f32 out =================
// grid = RP/64, block 128. A is [RP x lda] f16 row-major (zero-padded cols).
// WT0: [Cmid x lda] pre-swizzled, WT1: [Cout x Cmid] pre-swizzled.
__global__ void mlp2_kernel(const _Float16* __restrict__ A, int lda,
                            const _Float16* __restrict__ WT0, const float* __restrict__ b0,
                            int Cmid,
                            const _Float16* __restrict__ WT1, const float* __restrict__ b1,
                            int Cout, float* __restrict__ out, int reluLast) {
  __shared__ _Float16 h1[64 * 64];
  int w = threadIdx.x >> 5;
  size_t row0 = (size_t)blockIdx.x * 64 + w * 16;
  const _Float16* a0 = A + row0 * lda;
  _Float16* h1w = h1 + w * 16 * Cmid;
  for (int ct = 0; ct < Cmid; ct += 16) {
    v8f acc = {};
    for (int kc = 0; kc < lda; kc += 32) {
      v16h af = load_a_frag(a0, lda, kc);
      v16h bf = load_bt_frag(WT0, lda, kc, ct);
      acc = wmma_f16(af, bf, acc);
    }
    store_c_f16(acc, h1w, Cmid, ct, b0, true);
  }
  float* o0 = out + row0 * Cout;
  for (int ct = 0; ct < Cout; ct += 16) {
    v8f acc = {};
    for (int kc = 0; kc < Cmid; kc += 32) {
      v16h af = load_a_frag(h1w, Cmid, kc);
      v16h bf = load_bt_frag(WT1, Cmid, kc, ct);
      acc = wmma_f16(af, bf, acc);
    }
    store_c_f32(acc, o0, Cout, ct, b1, reluLast != 0);
  }
}

// ================= fused 6-layer tail: fp1(33->32->32->32) + head(32->32->32->32) ===========
__device__ __forceinline__ void layer32(const _Float16* src, _Float16* dst,
                                        const _Float16* WT, const float* bias, bool relu) {
  for (int ct = 0; ct < 32; ct += 16) {
    v8f acc = {};
    v16h af = load_a_frag(src, 32, 0);
    v16h bf = load_bt_frag(WT, 32, 0, ct);
    acc = wmma_f16(af, bf, acc);
    store_c_f16(acc, dst, 32, ct, bias, relu);
  }
}

__global__ void chain6_kernel(const _Float16* __restrict__ A,
                              const _Float16* W0, const float* B0, const _Float16* W1, const float* B1,
                              const _Float16* W2, const float* B2, const _Float16* W3, const float* B3,
                              const _Float16* W4, const float* B4, const _Float16* W5, const float* B5,
                              float* __restrict__ out) {
  __shared__ _Float16 bufA[64 * 32];
  __shared__ _Float16 bufB[64 * 32];
  int w = threadIdx.x >> 5;
  size_t row0 = (size_t)blockIdx.x * 64 + w * 16;
  _Float16* pa = bufA + w * 16 * 32;
  _Float16* pb = bufB + w * 16 * 32;
  // L0: [33 padded to 64] -> 32, relu
  for (int ct = 0; ct < 32; ct += 16) {
    v8f acc = {};
    for (int kc = 0; kc < 64; kc += 32) {
      v16h af = load_a_frag(A + row0 * 64, 64, kc);
      v16h bf = load_bt_frag(W0, 64, kc, ct);
      acc = wmma_f16(af, bf, acc);
    }
    store_c_f16(acc, pa, 32, ct, B0, true);
  }
  layer32(pa, pb, W1, B1, true);   // fp1 L1 relu
  layer32(pb, pa, W2, B2, false);  // fp1 L2 plain
  layer32(pa, pb, W3, B3, true);   // head L0 relu
  layer32(pb, pa, W4, B4, true);   // head L1 relu
  // head L2 plain -> f32 out
  float* o0 = out + row0 * OUTD;
  for (int ct = 0; ct < OUTD; ct += 16) {
    v8f acc = {};
    v16h af = load_a_frag(pa, 32, 0);
    v16h bf = load_bt_frag(W5, 32, 0, ct);
    acc = wmma_f16(af, bf, acc);
    store_c_f32(acc, o0, OUTD, ct, B5, false);
  }
}

// ================= small glue kernels =================
__global__ void concat_sa3_kernel(const float* __restrict__ x2, const float* __restrict__ pos2,
                                  _Float16* __restrict__ A3, int Rreal, int RP) {
  int e = blockIdx.x * 256 + threadIdx.x;
  if (e >= RP * 96) return;
  int row = e / 96, c = e - row * 96;
  float v = 0.0f;
  if (row < Rreal) {
    if (c < 64) v = x2[(size_t)row * 64 + c];
    else if (c < 67) v = pos2[(size_t)row * 3 + (c - 64)];
  }
  A3[e] = (_Float16)v;
}

__global__ void globalmax_kernel(const float* __restrict__ H3, float* __restrict__ x3, int M2) {
  int b = blockIdx.x, c = threadIdx.x;  // block = 64
  float mv = -__builtin_inff();
  for (int m = 0; m < M2; m++) mv = fmaxf(mv, H3[((size_t)b * M2 + m) * 64 + c]);
  x3[b * 64 + c] = mv;
}

__global__ void concat_fp3_kernel(const float* __restrict__ x3, const float* __restrict__ x2,
                                  _Float16* __restrict__ A4, int Rreal, int RP, int M2) {
  int e = blockIdx.x * 256 + threadIdx.x;
  if (e >= RP * 128) return;
  int row = e >> 7, c = e & 127;
  float v = 0.0f;
  if (row < Rreal) {
    if (c < 64) { int bb = row / M2; v = x3[bb * 64 + c]; }
    else        { v = x2[(size_t)row * 64 + (c - 64)]; }
  }
  A4[e] = (_Float16)v;
}

// 3-NN inverse-d2 interpolation + concat with skip features -> f16 MLP input rows
__global__ void interp_concat_kernel(const float* __restrict__ xsrc, const float* __restrict__ pos_src,
                                     const float* __restrict__ posq, const float* __restrict__ xskip,
                                     int Ms, int Nq, int C, int Fs, int lda,
                                     _Float16* __restrict__ Aout) {
  int b = blockIdx.y;
  int q = blockIdx.x * blockDim.x + threadIdx.x;
  if (q >= Nq) return;
  size_t qb = (size_t)b * Nq + q;
  float qx = posq[qb * 3 + 0], qy = posq[qb * 3 + 1], qz = posq[qb * 3 + 2];
  const float* P = pos_src + (size_t)b * Ms * 3;
  float d0 = __builtin_inff(), d1 = d0, d2v = d0;
  int i0 = 0, i1 = 0, i2 = 0;
  for (int i = 0; i < Ms; i++) {
    float dx = P[i * 3 + 0] - qx, dy = P[i * 3 + 1] - qy, dz = P[i * 3 + 2] - qz;
    float d = dx * dx + dy * dy + dz * dz;
    if (d < d0)      { d2v = d1; i2 = i1; d1 = d0; i1 = i0; d0 = d; i0 = i; }
    else if (d < d1) { d2v = d1; i2 = i1; d1 = d; i1 = i; }
    else if (d < d2v){ d2v = d; i2 = i; }
  }
  float w0 = 1.0f / fmaxf(d0, 1e-16f);
  float w1 = 1.0f / fmaxf(d1, 1e-16f);
  float w2 = 1.0f / fmaxf(d2v, 1e-16f);
  float ws = w0 + w1 + w2;
  const float* s0 = xsrc + ((size_t)b * Ms + i0) * C;
  const float* s1 = xsrc + ((size_t)b * Ms + i1) * C;
  const float* s2 = xsrc + ((size_t)b * Ms + i2) * C;
  _Float16* arow = Aout + qb * lda;
  for (int c = 0; c < C; c++) {
    float h = (w0 * s0[c] + w1 * s1[c] + w2 * s2[c]) / ws;
    arow[c] = (_Float16)h;
  }
  for (int c = 0; c < Fs; c++) arow[C + c] = (_Float16)xskip[qb * Fs + c];
  for (int c = C + Fs; c < lda; c++) arow[c] = (_Float16)0.0f;
}

__global__ void fillzero_f16(_Float16* p, int n) {
  int i = blockIdx.x * 256 + threadIdx.x;
  if (i < n) p[i] = (_Float16)0.0f;
}

// ================= host launcher =================
extern "C" void kernel_launch(void* const* d_in, const int* in_sizes, int n_in,
                              void* d_out, int out_size, void* d_ws, size_t ws_size,
                              hipStream_t stream) {
  (void)n_in; (void)out_size; (void)ws_size;
  const float* x   = (const float*)d_in[0];
  const float* pos = (const float*)d_in[1];

  // Resolve param leaf indices: pytree-sorted keys (fp1 first, W0 size 1056)
  // vs dict insertion order (sa1 first, W0 size 4*32=128).
  int sa1i, sa2i, sa3i, fp3i, fp2i, fp1i, mlpi;
  if (in_sizes[2] == 128) {           // insertion order: sa1,sa2,sa3,fp3,fp2,fp1,mlp
    sa1i = 2; sa2i = 6; sa3i = 10; fp3i = 14; fp2i = 18; fp1i = 22; mlpi = 28;
  } else {                             // sorted keys: fp1,fp2,fp3,mlp,sa1,sa2,sa3
    fp1i = 2; fp2i = 8; fp3i = 12; mlpi = 16; sa1i = 22; sa2i = 26; sa3i = 30;
  }
  const float* sa1W0 = (const float*)d_in[sa1i + 0]; const float* sa1b0 = (const float*)d_in[sa1i + 1];
  const float* sa1W1 = (const float*)d_in[sa1i + 2]; const float* sa1b1 = (const float*)d_in[sa1i + 3];
  const float* sa2W0 = (const float*)d_in[sa2i + 0]; const float* sa2b0 = (const float*)d_in[sa2i + 1];
  const float* sa2W1 = (const float*)d_in[sa2i + 2]; const float* sa2b1 = (const float*)d_in[sa2i + 3];
  const float* sa3W0 = (const float*)d_in[sa3i + 0]; const float* sa3b0 = (const float*)d_in[sa3i + 1];
  const float* sa3W1 = (const float*)d_in[sa3i + 2]; const float* sa3b1 = (const float*)d_in[sa3i + 3];
  const float* fp3W0 = (const float*)d_in[fp3i + 0]; const float* fp3b0 = (const float*)d_in[fp3i + 1];
  const float* fp3W1 = (const float*)d_in[fp3i + 2]; const float* fp3b1 = (const float*)d_in[fp3i + 3];
  const float* fp2W0 = (const float*)d_in[fp2i + 0]; const float* fp2b0 = (const float*)d_in[fp2i + 1];
  const float* fp2W1 = (const float*)d_in[fp2i + 2]; const float* fp2b1 = (const float*)d_in[fp2i + 3];
  const float* fp1W0 = (const float*)d_in[fp1i + 0]; const float* fp1b0 = (const float*)d_in[fp1i + 1];
  const float* fp1W1 = (const float*)d_in[fp1i + 2]; const float* fp1b1 = (const float*)d_in[fp1i + 3];
  const float* fp1W2 = (const float*)d_in[fp1i + 4]; const float* fp1b2 = (const float*)d_in[fp1i + 5];
  const float* mW0 = (const float*)d_in[mlpi + 0]; const float* mb0 = (const float*)d_in[mlpi + 1];
  const float* mW1 = (const float*)d_in[mlpi + 2]; const float* mb1 = (const float*)d_in[mlpi + 3];
  const float* mW2 = (const float*)d_in[mlpi + 4]; const float* mb2 = (const float*)d_in[mlpi + 5];

  // ---- workspace layout (256B aligned bump allocator) ----
  char* base = (char*)d_ws;
  size_t off = 0;
  auto take = [&](size_t bytes) -> char* {
    char* p = base + off;
    off += (bytes + 255) & ~(size_t)255;
    return p;
  };
  float*     pos1 = (float*)take((size_t)B_ * M1_ * 3 * 4);
  int*       nbr1 = (int*)take((size_t)B_ * M1_ * KNB * 4);
  float*     d2_1 = (float*)take((size_t)B_ * M1_ * KNB * 4);
  float*     x1   = (float*)take((size_t)B_ * M1_ * 32 * 4);
  float*     pos2 = (float*)take((size_t)B_ * M2_ * 3 * 4);
  int*       nbr2 = (int*)take((size_t)B_ * M2_ * KNB * 4);
  float*     d2_2 = (float*)take((size_t)B_ * M2_ * KNB * 4);
  float*     x2   = (float*)take((size_t)B_ * M2_ * 64 * 4);
  _Float16*  A3   = (_Float16*)take((size_t)RP3_ * 96 * 2);
  float*     H3   = (float*)take((size_t)RP3_ * 64 * 4);
  float*     x3   = (float*)take((size_t)B_ * 64 * 4);
  _Float16*  A4   = (_Float16*)take((size_t)RP3_ * 128 * 2);
  float*     xf3  = (float*)take((size_t)RP3_ * 64 * 4);
  _Float16*  A5   = (_Float16*)take((size_t)RP5_ * 96 * 2);
  float*     xf2  = (float*)take((size_t)RP5_ * 32 * 4);
  _Float16*  A6   = (_Float16*)take((size_t)R6_ * 64 * 2);
  // pre-swizzled f16 weights WT[Cout x KP]
  _Float16* wt_sa1_0 = (_Float16*)take(32 * 32 * 2);
  _Float16* wt_sa1_1 = (_Float16*)take(32 * 32 * 2);
  _Float16* wt_sa2_0 = (_Float16*)take(32 * 64 * 2);
  _Float16* wt_sa2_1 = (_Float16*)take(64 * 32 * 2);
  _Float16* wt_sa3_0 = (_Float16*)take(64 * 96 * 2);
  _Float16* wt_sa3_1 = (_Float16*)take(64 * 64 * 2);
  _Float16* wt_fp3_0 = (_Float16*)take(64 * 128 * 2);
  _Float16* wt_fp3_1 = (_Float16*)take(64 * 64 * 2);
  _Float16* wt_fp2_0 = (_Float16*)take(32 * 96 * 2);
  _Float16* wt_fp2_1 = (_Float16*)take(32 * 32 * 2);
  _Float16* wt_fp1_0 = (_Float16*)take(32 * 64 * 2);
  _Float16* wt_fp1_1 = (_Float16*)take(32 * 32 * 2);
  _Float16* wt_fp1_2 = (_Float16*)take(32 * 32 * 2);
  _Float16* wt_m0    = (_Float16*)take(32 * 32 * 2);
  _Float16* wt_m1    = (_Float16*)take(32 * 32 * 2);
  _Float16* wt_m2    = (_Float16*)take(32 * 32 * 2);
  float*    out  = (float*)d_out;

  // ---- weight pre-swizzle (tiny, amortized over ~18K WMMA workgroups) ----
#define PREP(W, Cin, Cout, KP, WT) \
  prep_wt_kernel<<<((Cout)*(KP) + 255) / 256, 256, 0, stream>>>((W), (Cin), (Cout), (KP), (WT))
  PREP(sa1W0,  4, 32, 32,  wt_sa1_0);
  PREP(sa1W1, 32, 32, 32,  wt_sa1_1);
  PREP(sa2W0, 35, 32, 64,  wt_sa2_0);
  PREP(sa2W1, 32, 64, 32,  wt_sa2_1);
  PREP(sa3W0, 67, 64, 96,  wt_sa3_0);
  PREP(sa3W1, 64, 64, 64,  wt_sa3_1);
  PREP(fp3W0, 128, 64, 128, wt_fp3_0);
  PREP(fp3W1, 64, 64, 64,  wt_fp3_1);
  PREP(fp2W0, 96, 32, 96,  wt_fp2_0);
  PREP(fp2W1, 32, 32, 32,  wt_fp2_1);
  PREP(fp1W0, 33, 32, 64,  wt_fp1_0);
  PREP(fp1W1, 32, 32, 32,  wt_fp1_1);
  PREP(fp1W2, 32, 32, 32,  wt_fp1_2);
  PREP(mW0,   32, 32, 32,  wt_m0);
  PREP(mW1,   32, 32, 32,  wt_m1);
  PREP(mW2,   32, 32, 32,  wt_m2);
#undef PREP

  // ---- stage 1: SA1 ----
  fps_kernel<<<B_, 256, 0, stream>>>(pos, N_, M1_, pos1);
  topk_kernel<<<dim3(M1_, B_), 256, 0, stream>>>(pos, N_, pos1, M1_, nbr1, d2_1);
  sa_fused_kernel<<<dim3(M1_, B_), 128, 0, stream>>>(x, pos, pos1, nbr1, d2_1, R1SQ,
                                                     N_, M1_, FEAT_, 32, 5, 32,
                                                     wt_sa1_0, sa1b0, wt_sa1_1, sa1b1, x1);
  // ---- stage 2: SA2 ----
  fps_kernel<<<B_, 256, 0, stream>>>(pos1, M1_, M2_, pos2);
  topk_kernel<<<dim3(M2_, B_), 256, 0, stream>>>(pos1, M1_, pos2, M2_, nbr2, d2_2);
  sa_fused_kernel<<<dim3(M2_, B_), 128, 0, stream>>>(x1, pos1, pos2, nbr2, d2_2, R2SQ,
                                                     M1_, M2_, 32, 64, 6, 64,
                                                     wt_sa2_0, sa2b0, wt_sa2_1, sa2b1, x2);
  // ---- global SA3 ----
  concat_sa3_kernel<<<(RP3_ * 96 + 255) / 256, 256, 0, stream>>>(x2, pos2, A3, R3_, RP3_);
  mlp2_kernel<<<RP3_ / 64, 128, 0, stream>>>(A3, 96, wt_sa3_0, sa3b0, 64,
                                             wt_sa3_1, sa3b1, 64, H3, 0);
  globalmax_kernel<<<B_, 64, 0, stream>>>(H3, x3, M2_);
  // ---- FP3 (k=1 from single origin source == broadcast x3) ----
  concat_fp3_kernel<<<(RP3_ * 128 + 255) / 256, 256, 0, stream>>>(x3, x2, A4, R3_, RP3_, M2_);
  mlp2_kernel<<<RP3_ / 64, 128, 0, stream>>>(A4, 128, wt_fp3_0, fp3b0, 64,
                                             wt_fp3_1, fp3b1, 64, xf3, 0);
  // ---- FP2: interp xf3 (pos2 -> pos1), concat x1 ----
  interp_concat_kernel<<<dim3((M1_ + 255) / 256, B_), 256, 0, stream>>>(
      xf3, pos2, pos1, x1, M2_, M1_, 64, 32, 96, A5);
  fillzero_f16<<<((RP5_ - R5_) * 96 + 255) / 256, 256, 0, stream>>>(A5 + (size_t)R5_ * 96,
                                                                    (RP5_ - R5_) * 96);
  mlp2_kernel<<<RP5_ / 64, 128, 0, stream>>>(A5, 96, wt_fp2_0, fp2b0, 32,
                                             wt_fp2_1, fp2b1, 32, xf2, 0);
  // ---- FP1: interp xf2 (pos1 -> pos), concat x ----
  interp_concat_kernel<<<dim3(N_ / 256, B_), 256, 0, stream>>>(
      xf2, pos1, pos, x, M1_, N_, 32, FEAT_, 64, A6);
  // ---- fused fp1 (3 layers) + output head (3 layers) -> d_out ----
  chain6_kernel<<<R6_ / 64, 128, 0, stream>>>(A6,
      wt_fp1_0, fp1b0, wt_fp1_1, fp1b1, wt_fp1_2, fp1b2,
      wt_m0, mb0, wt_m1, mb1, wt_m2, mb2, out);
}
